// GAT_3layer_17205638988375
// MI455X (gfx1250) — matrix operand order repaired and checked by reading, not compiled, not run on hardware
//
#include <hip/hip_runtime.h>
#include <hip/hip_bf16.h>
#include <math.h>

// ---------------------------------------------------------------------------
// Types for CDNA5 WMMA
// ---------------------------------------------------------------------------
typedef __bf16 v16bf __attribute__((ext_vector_type(16)));
typedef float  v8f   __attribute__((ext_vector_type(8)));

#define NEG_SLOPE 0.2f
#define HDIM 128           // feature width (F_in == H == 128)
#define GPOOL 128          // number of graphs
#define TILE_M 128         // rows per block in GEMM
#define LDS_PAD 136        // 128 + 8: keeps rows 16B-aligned, near-zero bank conflicts

// ---------------------------------------------------------------------------
// One-time per layer: Wt[n][k] = bf16(W[k][n])  (tiny, 16384 elements)
// ---------------------------------------------------------------------------
__global__ void transpose_w_bf16(const float* __restrict__ W, __bf16* __restrict__ Wt)
{
    int idx = blockIdx.x * blockDim.x + threadIdx.x;
    if (idx >= HDIM * HDIM) return;
    int k = idx >> 7, n = idx & 127;               // coalesced read, scattered 2B write (tiny)
    Wt[(size_t)n * HDIM + k] = (__bf16)W[idx];
}

// ---------------------------------------------------------------------------
// One-time: materialize int32 src/dst with self loops appended
// ---------------------------------------------------------------------------
__global__ void build_edges(const long long* __restrict__ ei, int E, int Etot,
                            int* __restrict__ s32, int* __restrict__ d32)
{
    int e = blockIdx.x * blockDim.x + threadIdx.x;
    if (e >= Etot) return;
    if (e < E) { s32[e] = (int)ei[e]; d32[e] = (int)ei[(size_t)E + e]; }
    else       { s32[e] = e - E;      d32[e] = e - E; }
}

// ---------------------------------------------------------------------------
// GEMM: H[N,128] = A[N,128] @ W[128,128]   (bf16 inputs, f32 accumulate, WMMA)
// block = 256 threads = 8 waves; each wave computes 16 rows x 128 cols.
// Wt is pre-transposed bf16 [n][k].
// ---------------------------------------------------------------------------
__global__ __launch_bounds__(256)
void gemm_bf16_wmma(const float* __restrict__ A, const __bf16* __restrict__ Wt,
                    float* __restrict__ Hout, int N)
{
    __shared__ __bf16 Alds[TILE_M][LDS_PAD];   // [row][k]
    __shared__ __bf16 Blds[HDIM][LDS_PAD];     // [n][k]

    const int tid     = threadIdx.x;
    const int rowBase = blockIdx.x * TILE_M;

    // --- stage A tile: float4 global loads -> packed 4xbf16 (8B) LDS stores ---
    #pragma unroll
    for (int i = 0; i < (TILE_M * HDIM / 4) / 256; ++i) {   // 16 iters
        int v  = tid + i * 256;
        int r  = v >> 5;                 // row
        int c4 = v & 31;                 // float4 index within row
        int grow = rowBase + r;
        float4 val = (grow < N) ? ((const float4*)(A + (size_t)grow * HDIM))[c4]
                                : make_float4(0.f, 0.f, 0.f, 0.f);
        union { __bf16 b[4]; uint2 u; } pk;
        pk.b[0] = (__bf16)val.x; pk.b[1] = (__bf16)val.y;
        pk.b[2] = (__bf16)val.z; pk.b[3] = (__bf16)val.w;
        *(uint2*)&Alds[r][c4 * 4] = pk.u;
    }
    // --- stage Wt (already bf16 [n][k]): 16B loads -> 16B LDS stores ---
    #pragma unroll
    for (int i = 0; i < (HDIM * HDIM / 8) / 256; ++i) {     // 8 iters
        int v  = tid + i * 256;
        int n  = v >> 4;
        int k8 = (v & 15) * 8;
        uint4 w = ((const uint4*)(Wt + (size_t)n * HDIM))[k8 >> 3];
        *(uint4*)&Blds[n][k8] = w;
    }
    __syncthreads();

    const int lane   = tid & 31;
    const int waveId = tid >> 5;
    const int lrow   = lane & 15;
    const int khalf  = lane >> 4;

    v8f acc[8];
    #pragma unroll
    for (int nt = 0; nt < 8; ++nt) acc[nt] = (v8f){0.f,0.f,0.f,0.f,0.f,0.f,0.f,0.f};

    const int arow = waveId * 16 + lrow;

    #pragma unroll
    for (int kb = 0; kb < 4; ++kb) {
        // A fragment: lane<16 -> K {0..7, 16..23}; lane>=16 -> K {8..15, 24..31}
        v16bf afrag;
        const int ka = kb * 32 + khalf * 8;
        #pragma unroll
        for (int i = 0; i < 8; ++i) afrag[i]     = Alds[arow][ka + i];
        #pragma unroll
        for (int i = 0; i < 8; ++i) afrag[8 + i] = Alds[arow][ka + 16 + i];

        #pragma unroll
        for (int nt = 0; nt < 8; ++nt) {
            // B fragment: lane<16 -> K {0..7, 16..23}; lane>=16 -> K {8..15, 24..31}
            v16bf bfrag;
            const int n  = nt * 16 + lrow;
            const int kB = kb * 32 + khalf * 8;
            #pragma unroll
            for (int i = 0; i < 16; ++i) bfrag[i] = Blds[n][kB + (i < 8 ? i : 8 + i)];

            acc[nt] = __builtin_amdgcn_wmma_f32_16x16x32_bf16(
                false, afrag, false, bfrag, (short)0, acc[nt], false, false);
        }
    }

    // --- write back: C layout: VGPR r -> row (khalf*8 + r), col = lane&15 ---
    const int outRow0 = rowBase + waveId * 16 + khalf * 8;
    #pragma unroll
    for (int nt = 0; nt < 8; ++nt) {
        const int col = nt * 16 + lrow;
        #pragma unroll
        for (int r = 0; r < 8; ++r) {
            const int grow = outRow0 + r;
            if (grow < N) Hout[(size_t)grow * HDIM + col] = acc[nt][r];
        }
    }
}

// ---------------------------------------------------------------------------
// Per-node attention coefficients: as[i] = h[i].a_src, ad[i] = h[i].a_dst
// one wave per node; float4 per lane; wave32 shfl reduction.
// ---------------------------------------------------------------------------
__global__ __launch_bounds__(256)
void alphas_kernel(const float* __restrict__ h,
                   const float* __restrict__ a_src, const float* __restrict__ a_dst,
                   float* __restrict__ as, float* __restrict__ ad, int N)
{
    int node = blockIdx.x * (blockDim.x >> 5) + (threadIdx.x >> 5);
    if (node >= N) return;
    int lane = threadIdx.x & 31;
    float4 hv = ((const float4*)(h + (size_t)node * HDIM))[lane];
    float4 sv = ((const float4*)a_src)[lane];
    float4 dv = ((const float4*)a_dst)[lane];
    float ps = hv.x*sv.x + hv.y*sv.y + hv.z*sv.z + hv.w*sv.w;
    float pd = hv.x*dv.x + hv.y*dv.y + hv.z*dv.z + hv.w*dv.w;
    #pragma unroll
    for (int off = 16; off > 0; off >>= 1) {
        ps += __shfl_down(ps, off, 32);
        pd += __shfl_down(pd, off, 32);
    }
    if (lane == 0) { as[node] = ps; ad[node] = pd; }
}

// ---------------------------------------------------------------------------
// Order-preserving float <-> uint encoding for atomic max
// ---------------------------------------------------------------------------
__device__ __forceinline__ unsigned int fenc(float f) {
    unsigned int b = __float_as_uint(f);
    return (b & 0x80000000u) ? ~b : (b | 0x80000000u);
}
__device__ __forceinline__ float fdec(unsigned int u) {
    unsigned int b = (u & 0x80000000u) ? (u & 0x7FFFFFFFu) : ~u;
    return __uint_as_float(b);
}

__global__ void fill_f32(float* p, float v, size_t n) {
    size_t i = (size_t)blockIdx.x * blockDim.x + threadIdx.x;
    if (i < n) p[i] = v;
}
__global__ void fill_u32(unsigned int* p, unsigned int v, size_t n) {
    size_t i = (size_t)blockIdx.x * blockDim.x + threadIdx.x;
    if (i < n) p[i] = v;
}

// logits + leaky relu + segment max (encoded uint atomicMax)
__global__ void edge_logits_kernel(const float* __restrict__ as, const float* __restrict__ ad,
                                   const int* __restrict__ src, const int* __restrict__ dst,
                                   int Etot,
                                   float* __restrict__ logits, unsigned int* __restrict__ mEnc)
{
    int e = blockIdx.x * blockDim.x + threadIdx.x;
    if (e >= Etot) return;
    int s = src[e], d = dst[e];
    float l = as[s] + ad[d];
    l = (l > 0.0f) ? l : NEG_SLOPE * l;
    logits[e] = l;
    atomicMax(mEnc + d, fenc(l));
}

__global__ void decode_m_kernel(unsigned int* m, int N) {
    int i = blockIdx.x * blockDim.x + threadIdx.x;
    if (i < N) {
        float v = fdec(m[i]);
        ((float*)m)[i] = v;
    }
}

// e = exp(logit - m[dst]); denom[dst] += e   (ew written in place over logits)
__global__ void edge_exp_kernel(float* __restrict__ logits, const float* __restrict__ m,
                                const int* __restrict__ dst, int Etot,
                                float* __restrict__ denom)
{
    int e = blockIdx.x * blockDim.x + threadIdx.x;
    if (e >= Etot) return;
    int d = dst[e];
    float w = __expf(logits[e] - m[d]);
    logits[e] = w;
    atomicAdd(denom + d, w);
}

// out[dst] += h[src] * (ew/denom[dst]); one wave per edge, float4 per lane
__global__ __launch_bounds__(256)
void scatter_kernel(const float* __restrict__ h, const float* __restrict__ ew,
                    const float* __restrict__ denom,
                    const int* __restrict__ src, const int* __restrict__ dst,
                    int Etot, float* __restrict__ out)
{
    int e = blockIdx.x * (blockDim.x >> 5) + (threadIdx.x >> 5);
    if (e >= Etot) return;
    int lane = threadIdx.x & 31;
    int s = src[e], d = dst[e];
    float w = ew[e] / denom[d];
    float4 hv = ((const float4*)(h + (size_t)s * HDIM))[lane];
    float* o = out + (size_t)d * HDIM + lane * 4;
    atomicAdd(o + 0, hv.x * w);
    atomicAdd(o + 1, hv.y * w);
    atomicAdd(o + 2, hv.z * w);
    atomicAdd(o + 3, hv.w * w);
}

__global__ void bias_relu_kernel(float* __restrict__ buf, const float* __restrict__ b, size_t n) {
    size_t i = (size_t)blockIdx.x * blockDim.x + threadIdx.x;
    if (i < n) {
        float v = buf[i] + b[i & (HDIM - 1)];
        buf[i] = v > 0.0f ? v : 0.0f;
    }
}

// global mean pool accumulation: one block (128 threads) per node
__global__ __launch_bounds__(128)
void pool_kernel(const float* __restrict__ h, const long long* __restrict__ batch,
                 float* __restrict__ sums, float* __restrict__ cnt, int N)
{
    int node = blockIdx.x;
    if (node >= N) return;
    long long g = batch[node];
    atomicAdd(sums + (size_t)g * HDIM + threadIdx.x, h[(size_t)node * HDIM + threadIdx.x]);
    if (threadIdx.x == 0) atomicAdd(cnt + g, 1.0f);
}

// out[g,c] = (sums[g]/max(cnt,1)) @ lin_W + lin_b   -- tiny 128x128x10
__global__ __launch_bounds__(64)
void final_linear_kernel(const float* __restrict__ sums, const float* __restrict__ cnt,
                         const float* __restrict__ lin_W, const float* __restrict__ lin_b,
                         float* __restrict__ out, int C)
{
    int g = blockIdx.x;
    int c = threadIdx.x;
    if (c >= C) return;
    float inv = 1.0f / fmaxf(cnt[g], 1.0f);
    float acc = lin_b[c];
    #pragma unroll 8
    for (int k = 0; k < HDIM; ++k)
        acc += (sums[(size_t)g * HDIM + k] * inv) * lin_W[(size_t)k * C + c];
    out[(size_t)g * C + c] = acc;
}

// ---------------------------------------------------------------------------
// Host launcher (all launches on `stream`; graph-capture safe)
// ---------------------------------------------------------------------------
extern "C" void kernel_launch(void* const* d_in, const int* in_sizes, int n_in,
                              void* d_out, int out_size, void* d_ws, size_t ws_size,
                              hipStream_t stream)
{
    const float*      x     = (const float*)d_in[0];
    const long long*  ei    = (const long long*)d_in[1];
    const long long*  batch = (const long long*)d_in[2];
    const float* Ws[3] = { (const float*)d_in[3], (const float*)d_in[7],  (const float*)d_in[11] };
    const float* As[3] = { (const float*)d_in[4], (const float*)d_in[8],  (const float*)d_in[12] };
    const float* Ad[3] = { (const float*)d_in[5], (const float*)d_in[9],  (const float*)d_in[13] };
    const float* Bs[3] = { (const float*)d_in[6], (const float*)d_in[10], (const float*)d_in[14] };
    const float* linW = (const float*)d_in[15];
    const float* linB = (const float*)d_in[16];
    float* out = (float*)d_out;

    const int N    = in_sizes[0] / HDIM;
    const int E    = in_sizes[1] / 2;
    const int Etot = E + N;                  // self loops appended
    const int C    = in_sizes[16];           // 10

    // workspace carve-up (256B aligned)
    char* p = (char*)d_ws;
    auto walloc = [&](size_t bytes) -> void* {
        void* r = (void*)p;
        p += (bytes + 255) & ~(size_t)255;
        return r;
    };
    float*        hA     = (float*)walloc((size_t)N * HDIM * 4);
    float*        hB     = (float*)walloc((size_t)N * HDIM * 4);
    float*        alphaS = (float*)walloc((size_t)N * 4);
    float*        alphaD = (float*)walloc((size_t)N * 4);
    unsigned int* mEnc   = (unsigned int*)walloc((size_t)N * 4);
    float*        denom  = (float*)walloc((size_t)N * 4);
    float*        ewBuf  = (float*)walloc((size_t)Etot * 4);
    int*          src32  = (int*)walloc((size_t)Etot * 4);
    int*          dst32  = (int*)walloc((size_t)Etot * 4);
    __bf16*       Wt     = (__bf16*)walloc((size_t)HDIM * HDIM * 2);
    float*        sums   = (float*)walloc((size_t)GPOOL * HDIM * 4);
    float*        cnt    = (float*)walloc((size_t)GPOOL * 4);

    const int gemmGrid  = (N + TILE_M - 1) / TILE_M;
    const int nodeWaves = (N + 7) / 8;
    const int edgeThr   = (Etot + 255) / 256;
    const int edgeWaves = (Etot + 7) / 8;
    const int nThr      = (N + 255) / 256;
    const size_t NH     = (size_t)N * HDIM;
    const int nhThr     = (int)((NH + 255) / 256);

    // encoded(-inf): bits(-inf)=0xFF800000 (negative) -> ~bits = 0x007FFFFF
    const unsigned int ENC_NINF = 0x007FFFFFu;

    build_edges<<<edgeThr, 256, 0, stream>>>(ei, E, Etot, src32, dst32);

    const float* feat = x;
    for (int l = 0; l < 3; ++l) {
        transpose_w_bf16<<<(HDIM * HDIM + 255) / 256, 256, 0, stream>>>(Ws[l], Wt);
        gemm_bf16_wmma<<<gemmGrid, 256, 0, stream>>>(feat, Wt, hA, N);
        alphas_kernel<<<nodeWaves, 256, 0, stream>>>(hA, As[l], Ad[l], alphaS, alphaD, N);
        fill_u32<<<nThr, 256, 0, stream>>>(mEnc, ENC_NINF, (size_t)N);
        edge_logits_kernel<<<edgeThr, 256, 0, stream>>>(alphaS, alphaD, src32, dst32, Etot, ewBuf, mEnc);
        decode_m_kernel<<<nThr, 256, 0, stream>>>(mEnc, N);
        fill_f32<<<nThr, 256, 0, stream>>>(denom, 0.0f, (size_t)N);
        edge_exp_kernel<<<edgeThr, 256, 0, stream>>>(ewBuf, (const float*)mEnc, dst32, Etot, denom);
        fill_f32<<<nhThr, 256, 0, stream>>>(hB, 0.0f, NH);
        scatter_kernel<<<edgeWaves, 256, 0, stream>>>(hA, ewBuf, denom, src32, dst32, Etot, hB);
        bias_relu_kernel<<<nhThr, 256, 0, stream>>>(hB, Bs[l], NH);
        feat = hB;
    }

    fill_f32<<<(GPOOL * HDIM + 255) / 256, 256, 0, stream>>>(sums, 0.0f, (size_t)GPOOL * HDIM);
    fill_f32<<<1, 256, 0, stream>>>(cnt, 0.0f, (size_t)GPOOL);
    pool_kernel<<<N, 128, 0, stream>>>(hB, batch, sums, cnt, N);
    final_linear_kernel<<<GPOOL, 64, 0, stream>>>(sums, cnt, linW, linB, out, C);
}
